// SABlock_78520592105699
// MI455X (gfx1250) — matrix operand order repaired and checked by reading, not compiled, not run on hardware
//
#include <hip/hip_runtime.h>
#include <hip/hip_bf16.h>

// Problem constants (match reference): B=4, C=512, CI=256, N=H*W=4096
#define NB 4
#define CC 512
#define CIc 256
#define NN 4096

typedef __bf16 bf16_t;
typedef __attribute__((ext_vector_type(16))) __bf16 v16bf;
typedef __attribute__((ext_vector_type(8)))  float  v8f;

// float -> bf16, round-to-nearest-even
__device__ __forceinline__ bf16_t f2bf(float f) {
  unsigned u = __builtin_bit_cast(unsigned, f);
  u += 0x7fffu + ((u >> 16) & 1u);
  unsigned short h = (unsigned short)(u >> 16);
  return __builtin_bit_cast(bf16_t, h);
}

// ---------------- WMMA fragment helpers (wave32, 16x16x32 bf16) -------------
// A-matrix 16x32 (MxK), bf16, per ISA table:
//   lane = {half(=lane>>4), m(=lane&15)}; VGPR0-3: K = half*8 + {0..7};
//   VGPR4-7: K = 16 + half*8 + {0..7}. Memory layout: row-major, K contiguous.
__device__ __forceinline__ v16bf load_fragA(const bf16_t* base,
                                            int ld, int row_base, int k_base) {
  const int lane = threadIdx.x & 31;
  const int row = lane & 15, half = lane >> 4;
  const bf16_t* p = base + (size_t)(row_base + row) * ld + k_base + half * 8;
  union { v16bf f; uint4 q[2]; } u;
  u.q[0] = *reinterpret_cast<const uint4*>(p);        // K = kb+half*8 .. +7
  u.q[1] = *reinterpret_cast<const uint4*>(p + 16);   // K = kb+16+half*8 .. +7
  return u.f;
}

// B-matrix 32x16 (KxN), bf16: same K order as A: VGPR0-3: K = half*8 + {0..7};
// VGPR4-7: K = 16 + half*8 + {0..7}.  Memory layout: column(n)-major, K contiguous.
__device__ __forceinline__ v16bf load_fragB(const bf16_t* base,
                                            int ld, int col_base, int k_base) {
  const int lane = threadIdx.x & 31;
  const int col = lane & 15, half = lane >> 4;
  const bf16_t* p = base + (size_t)(col_base + col) * ld + k_base + half * 8;
  union { v16bf f; uint4 q[2]; } u;
  u.q[0] = *reinterpret_cast<const uint4*>(p);        // K = kb+half*8 .. +7
  u.q[1] = *reinterpret_cast<const uint4*>(p + 16);   // K = kb+16+half*8 .. +7
  return u.f;
}

__device__ __forceinline__ v8f wmma_bf16(v16bf a, v16bf b, v8f c) {
  return __builtin_amdgcn_wmma_f32_16x16x32_bf16(
      /*neg_a=*/false, a, /*neg_b=*/false, b,
      /*c_mod=*/(short)0, c, /*reuse_a=*/false, /*reuse_b=*/false);
}

// Row reductions across the 16 lanes of a lane-half (C/D layout: row group is
// selected by lane>>4, columns by lane&15).
__device__ __forceinline__ float red_max16(float v) {
#pragma unroll
  for (int m = 1; m < 16; m <<= 1) v = fmaxf(v, __shfl_xor(v, m, 32));
  return v;
}
__device__ __forceinline__ float red_sum16(float v) {
#pragma unroll
  for (int m = 1; m < 16; m <<= 1) v += __shfl_xor(v, m, 32);
  return v;
}

// One lane's 16B async DMA: global -> LDS, tracked by ASYNCcnt.
__device__ __forceinline__ void async_b128(void* lds_dst, const bf16_t* gsrc) {
  unsigned l = (unsigned)(size_t)lds_dst;  // low 32 bits of a shared ptr = LDS offset
  asm volatile("global_load_async_to_lds_b128 %0, %1, off"
               :: "v"(l), "v"(gsrc) : "memory");
}

// ---------------- 1) layout/precision conversion ---------------------------
__global__ __launch_bounds__(256) void convert_x_kernel(
    const float* __restrict__ x, bf16_t* __restrict__ xt) {
  size_t idx = (size_t)blockIdx.x * blockDim.x + threadIdx.x;
  if (idx >= (size_t)NB * CC * NN) return;
  int n = (int)(idx % NN);
  int c = (int)((idx / NN) % CC);
  int b = (int)(idx / ((size_t)CC * NN));
  xt[((size_t)b * NN + n) * CC + c] = f2bf(x[idx]);
}

__global__ __launch_bounds__(256) void convert_w_kernel(
    const float* __restrict__ wphi, const float* __restrict__ wtheta,
    const float* __restrict__ wg, const float* __restrict__ wmask,
    bf16_t* __restrict__ o0, bf16_t* __restrict__ o1,
    bf16_t* __restrict__ o2, bf16_t* __restrict__ o3) {
  size_t idx = (size_t)blockIdx.x * blockDim.x + threadIdx.x;
  const size_t nper = (size_t)CIc * CC;
  if (idx >= 4 * nper) return;
  int which = (int)(idx / nper);
  size_t i = idx % nper;
  const float* s = which == 0 ? wphi : which == 1 ? wtheta : which == 2 ? wg : wmask;
  bf16_t* d = which == 0 ? o0 : which == 1 ? o1 : which == 2 ? o2 : o3;
  d[i] = f2bf(s[i]);
}

// ---------------- 2) projections: theta/phi -> (N,CI), g -> (CI,N) ---------
__global__ __launch_bounds__(256) void proj_kernel(
    const bf16_t* __restrict__ xt,
    const bf16_t* __restrict__ wphi, const bf16_t* __restrict__ wtheta,
    const bf16_t* __restrict__ wg,
    bf16_t* __restrict__ phi_t, bf16_t* __restrict__ theta_t,
    bf16_t* __restrict__ g_b) {
  const int wave = threadIdx.x >> 5;
  const int lane = threadIdx.x & 31;
  const int b = blockIdx.y;
  const int which = blockIdx.z;
  const int tile = blockIdx.x * 8 + wave;        // (N/16)*(CI/16) = 4096 tiles
  const int n0 = (tile >> 4) * 16;
  const int o0 = (tile & 15) * 16;
  const bf16_t* X = xt + (size_t)b * NN * CC;
  const bf16_t* W = which == 0 ? wphi : which == 1 ? wtheta : wg;

  v8f acc = {0.f, 0.f, 0.f, 0.f, 0.f, 0.f, 0.f, 0.f};
#pragma unroll
  for (int k = 0; k < CC; k += 32) {
    v16bf a = load_fragA(X, CC, n0, k);
    v16bf bb = load_fragB(W, CC, o0, k);
    acc = wmma_bf16(a, bb, acc);
  }
  const int colw = lane & 15, rh = (lane >> 4) * 8;
  if (which < 2) {                                // (N, CI) layout
    bf16_t* dst = (which == 0 ? phi_t : theta_t) + (size_t)b * NN * CIc;
#pragma unroll
    for (int r = 0; r < 8; ++r)
      dst[(size_t)(n0 + rh + r) * CIc + o0 + colw] = f2bf(acc[r]);
  } else {                                        // g: (CI, N) layout
    bf16_t* dst = g_b + (size_t)b * CIc * NN;
#pragma unroll
    for (int r = 0; r < 8; ++r)
      dst[(size_t)(o0 + colw) * NN + n0 + rh + r] = f2bf(acc[r]);
  }
}

// ---------------- 3) fused flash attention ---------------------------------
// Block = 4 waves = 64 query rows.  Per 32-key-column iteration the block
// stages phi(32x256) and g(256x32) tiles into LDS ONCE via async DMA
// (double-buffered, ASYNCcnt-tracked), then each wave runs 16 S-WMMAs,
// online softmax, and 16 O-WMMAs reading B-fragments from LDS.
__global__ __launch_bounds__(128, 1) void attn_kernel(
    const bf16_t* __restrict__ theta_t, const bf16_t* __restrict__ phi_t,
    const bf16_t* __restrict__ g_b, bf16_t* __restrict__ y_t) {
  // padded strides (264, 40 elems) -> each b128 frag read is bank-conflict-free
  __shared__ __align__(16) bf16_t phiS[2][32 * 264];   // [m_rel][k]
  __shared__ __align__(16) bf16_t gS[2][256 * 40];     // [ci][m_rel]
  __shared__ __align__(16) bf16_t plds[4][16 * 40];    // per-wave P staging

  const int tid = threadIdx.x;
  const int wave = tid >> 5;
  const int lane = tid & 31;
  const int b = blockIdx.y;
  const int n_base = blockIdx.x * 64 + wave * 16;
  const bf16_t* th = theta_t + (size_t)b * NN * CIc;
  const bf16_t* ph = phi_t + (size_t)b * NN * CIc;
  const bf16_t* gg = g_b + (size_t)b * CIc * NN;
  bf16_t* pl = &plds[wave][0];
  const int colw = lane & 15;
  const int rh = (lane >> 4) * 8;

  // theta fragments for our 16 rows stay resident (16 x 256 bf16 = 64 VGPRs)
  v16bf thF[8];
#pragma unroll
  for (int kt = 0; kt < 8; ++kt) thF[kt] = load_fragA(th, CIc, n_base, kt * 32);

  v8f O[16];
#pragma unroll
  for (int t = 0; t < 16; ++t)
#pragma unroll
    for (int r = 0; r < 8; ++r) O[t][r] = 0.f;

  float mrow[8], lrow[8];
#pragma unroll
  for (int r = 0; r < 8; ++r) { mrow[r] = -__builtin_inff(); lrow[r] = 0.f; }

  // stage one 32-column tile: 16 async b128 ops per wave (8 phi + 8 g)
  auto stage_tile = [&](int mb, int buf) {
#pragma unroll
    for (int i = 0; i < 8; ++i) {                 // phi: 32 rows x 256 k
      int e = (tid + i * 128) * 8;
      int mr = e >> 8, k = e & 255;
      async_b128(&phiS[buf][mr * 264 + k], ph + (size_t)(mb + mr) * CIc + k);
    }
#pragma unroll
    for (int i = 0; i < 8; ++i) {                 // g: 256 ci x 32 m
      int e = (tid + i * 128) * 8;
      int ci = e >> 5, mr = e & 31;
      async_b128(&gS[buf][ci * 40 + mr], gg + (size_t)ci * NN + mb + mr);
    }
  };

  stage_tile(0, 0);
  const int NIT = NN / 32;
  for (int it = 0; it < NIT; ++it) {
    const int mb = it * 32;
    const int buf = it & 1;
    const bool pre = (it + 1 < NIT);
    if (pre) {
      stage_tile(mb + 32, buf ^ 1);               // overlap next tile's DMA
      asm volatile("s_wait_asynccnt 0x10" ::: "memory");  // tile `it` done
    } else {
      asm volatile("s_wait_asynccnt 0x0" ::: "memory");
    }
    __syncthreads();                              // tile visible to all waves

    // scores S = theta^T @ phi for 32 key columns (B-frags from LDS)
    v8f S0 = {0.f, 0.f, 0.f, 0.f, 0.f, 0.f, 0.f, 0.f};
    v8f S1 = {0.f, 0.f, 0.f, 0.f, 0.f, 0.f, 0.f, 0.f};
#pragma unroll
    for (int kt = 0; kt < 8; ++kt) {
      v16bf bp0 = load_fragB(&phiS[buf][0], 264, 0, kt * 32);
      v16bf bp1 = load_fragB(&phiS[buf][0], 264, 16, kt * 32);
      S0 = wmma_bf16(thF[kt], bp0, S0);
      S1 = wmma_bf16(thF[kt], bp1, S1);
    }

    // online softmax; per-lane slot r <-> row (rh + r), same mapping as O
    float scale[8];
#pragma unroll
    for (int r = 0; r < 8; ++r) {
      float tm = red_max16(fmaxf(S0[r], S1[r]));
      float mn = fmaxf(mrow[r], tm);
      scale[r] = __expf(mrow[r] - mn);
      float p0 = __expf(S0[r] - mn);
      float p1 = __expf(S1[r] - mn);
      mrow[r] = mn;
      lrow[r] = lrow[r] * scale[r] + red_sum16(p0 + p1);
      pl[(rh + r) * 40 + colw]      = f2bf(p0);   // P -> LDS, row-major 16x32
      pl[(rh + r) * 40 + 16 + colw] = f2bf(p1);
    }
#pragma unroll
    for (int t = 0; t < 16; ++t)
#pragma unroll
      for (int r = 0; r < 8; ++r) O[t][r] *= scale[r];

    // relayout P (C/D layout) -> A fragment via LDS (same-wave, DS in order)
    asm volatile("s_wait_dscnt 0" ::: "memory");
    v16bf pf = load_fragA(pl, 40, 0, 0);

    // O(16 x CI) += P(16x32) @ G_t(32 x CI), B-frags from LDS
#pragma unroll
    for (int t = 0; t < 16; ++t) {
      v16bf gf = load_fragB(&gS[buf][0], 40, t * 16, 0);
      O[t] = wmma_bf16(pf, gf, O[t]);
    }
    __syncthreads();  // all waves done reading buf before it is re-filled
  }

  // normalize and store y_t in (N, CI) bf16
  bf16_t* yt = y_t + (size_t)b * NN * CIc;
#pragma unroll
  for (int r = 0; r < 8; ++r) lrow[r] = 1.f / lrow[r];
#pragma unroll
  for (int t = 0; t < 16; ++t)
#pragma unroll
    for (int r = 0; r < 8; ++r)
      yt[(size_t)(n_base + rh + r) * CIc + t * 16 + colw] =
          f2bf(O[t][r] * lrow[r]);
}

// ---------------- 4) mask projection: out(b,C,N) = Wm(C,CI) @ y ------------
__global__ __launch_bounds__(256) void mask_kernel(
    const bf16_t* __restrict__ y_t, const bf16_t* __restrict__ wm,
    float* __restrict__ out) {
  const int wave = threadIdx.x >> 5;
  const int lane = threadIdx.x & 31;
  const int b = blockIdx.y;
  const int tile = blockIdx.x * 8 + wave;     // (N/16)*(C/16) = 8192 tiles
  const int n0 = (tile >> 5) * 16;
  const int o0 = (tile & 31) * 16;
  const bf16_t* Y = y_t + (size_t)b * NN * CIc;

  v8f acc = {0.f, 0.f, 0.f, 0.f, 0.f, 0.f, 0.f, 0.f};
#pragma unroll
  for (int k = 0; k < CIc; k += 32) {
    v16bf a = load_fragA(Y, CIc, n0, k);
    v16bf bb = load_fragB(wm, CIc, o0, k);
    acc = wmma_bf16(a, bb, acc);
  }
  const int colw = lane & 15, rh = (lane >> 4) * 8;
  float* dst = out + (size_t)b * CC * NN;
#pragma unroll
  for (int r = 0; r < 8; ++r)
    dst[(size_t)(o0 + colw) * NN + n0 + rh + r] = acc[r];  // out[b, o, n]
}

// ---------------- launch ----------------------------------------------------
extern "C" void kernel_launch(void* const* d_in, const int* in_sizes, int n_in,
                              void* d_out, int out_size, void* d_ws,
                              size_t ws_size, hipStream_t stream) {
  const float* x       = (const float*)d_in[0];
  const float* w_phi   = (const float*)d_in[1];
  const float* w_theta = (const float*)d_in[2];
  const float* w_g     = (const float*)d_in[3];
  const float* w_mask  = (const float*)d_in[4];
  float* out = (float*)d_out;

  // workspace layout (~51.4 MB, all chunks 256B aligned)
  char* ws = (char*)d_ws;
  size_t off = 0;
  bf16_t* xt = (bf16_t*)(ws + off);       off += (size_t)NB * NN * CC * 2;
  bf16_t* wphi_b = (bf16_t*)(ws + off);   off += (size_t)CIc * CC * 2;
  bf16_t* wtheta_b = (bf16_t*)(ws + off); off += (size_t)CIc * CC * 2;
  bf16_t* wg_b = (bf16_t*)(ws + off);     off += (size_t)CIc * CC * 2;
  bf16_t* wmask_b = (bf16_t*)(ws + off);  off += (size_t)CC * CIc * 2;
  bf16_t* phi_t = (bf16_t*)(ws + off);    off += (size_t)NB * NN * CIc * 2;
  bf16_t* theta_t = (bf16_t*)(ws + off);  off += (size_t)NB * NN * CIc * 2;
  bf16_t* g_b = (bf16_t*)(ws + off);      off += (size_t)NB * CIc * NN * 2;
  bf16_t* y_t = (bf16_t*)(ws + off);      off += (size_t)NB * NN * CIc * 2;

  convert_x_kernel<<<(NB * CC * NN) / 256, 256, 0, stream>>>(x, xt);
  convert_w_kernel<<<(4 * CIc * CC) / 256, 256, 0, stream>>>(
      w_phi, w_theta, w_g, w_mask, wphi_b, wtheta_b, wg_b, wmask_b);

  proj_kernel<<<dim3(512, NB, 3), 256, 0, stream>>>(
      xt, wphi_b, wtheta_b, wg_b, phi_t, theta_t, g_b);

  attn_kernel<<<dim3(NN / 64, NB), 128, 0, stream>>>(theta_t, phi_t, g_b, y_t);

  mask_kernel<<<dim3(1024, NB), 256, 0, stream>>>(y_t, wmask_b, out);
}
